// LSTMModel_32083405701446
// MI455X (gfx1250) — compile-verified, matrix operand order checked by reading, not run on hardware
//
#include <hip/hip_runtime.h>
#include <hip/hip_bf16.h>

// ---------------------------------------------------------------------------
// LSTMP (proj LSTM) for MI455X / gfx1250: bf16 WMMA GEMMs, f32 accumulate.
// P=128, H=512, L=6, B=512, T=512, STEPS=32.
// ---------------------------------------------------------------------------

typedef __attribute__((ext_vector_type(16))) __bf16 v16bf;
typedef __attribute__((ext_vector_type(8)))  float  v8f;

union Frag {
    v16bf v;
    uint4 q[2];
};

#define NP    128        // proj size
#define NH    512        // hidden
#define NG    2048       // 4*H
#define NB    512        // batch
#define NT    512        // seq len
#define NL    6
#define NSTEP 32

// ---- workspace layout (bytes) ----
#define OFF_SEQ   0UL                       // [T][B][P] bf16          67,108,864
#define OFF_WG    67108864UL                // packed gates W, 6 MB
#define OFF_WHR   73400320UL                // packed Whr, 768 KB
#define OFF_WFC   74186752UL                // packed Wfc, 32 KB
#define OFF_BIAS  74219520UL                // b_ih+b_hh f32, 48 KB
#define OFF_H     74268672UL                // h_state [L][B][P] bf16, 768 KB
#define OFF_C     75055104UL                // c_state packed f32, 6 MB
#define OFF_ZERO  81346560UL                // zero bf16 [B][P], 128 KB
// zero-init region: OFF_H .. end = 7,208,960 bytes = 450,560 uint4

__device__ __forceinline__ float sigf(float x) {
    return 1.0f / (1.0f + __expf(-x));
}
__device__ __forceinline__ float tanh_f(float x) {
    float e = __expf(-2.0f * x);
    return (1.0f - e) / (1.0f + e);
}

// K offset inside a 16x32 bf16 fragment for (vgpr v, pair, laneHi) per ISA layout
__device__ __forceinline__ int k_off(int v, int pair, int laneHi) {
    int kk = (v < 4) ? (2 * v) : (16 + 2 * (v - 4));
    return (laneHi ? 8 : 0) + kk + pair;
}

// ---------------------------------------------------------------------------
// Weight repacking into B-fragment order: [frag][lane 0..31][16 bf16]
// ---------------------------------------------------------------------------
__global__ void pack_gates_k(const float* __restrict__ Wih,
                             const float* __restrict__ Whh,
                             __bf16* __restrict__ out) {
    long idx = (long)blockIdx.x * 256 + threadIdx.x;   // exactly 3,145,728
    int ev    = (int)(idx & 15);  long t = idx >> 4;
    int lane  = (int)(t & 31);    t >>= 5;
    int kf    = (int)(t & 7);     t >>= 3;
    int ntile = (int)(t & 127);   t >>= 7;
    int l     = (int)t;
    int v = ev >> 1, pair = ev & 1, hi = lane >> 4;
    int n  = ntile * 16 + (lane & 15);
    int ko = k_off(v, pair, hi);
    float val;
    if (kf < 4) val = Wih[((long)l * NG + n) * NP + kf * 32 + ko];
    else        val = Whh[((long)l * NG + n) * NP + (kf - 4) * 32 + ko];
    out[idx] = (__bf16)val;
}

__global__ void pack_whr_k(const float* __restrict__ Whr, __bf16* __restrict__ out) {
    long idx = (long)blockIdx.x * 256 + threadIdx.x;   // exactly 393,216
    int ev    = (int)(idx & 15);  long t = idx >> 4;
    int lane  = (int)(t & 31);    t >>= 5;
    int kf    = (int)(t & 15);    t >>= 4;
    int ntile = (int)(t & 7);     t >>= 3;
    int l     = (int)t;
    int v = ev >> 1, pair = ev & 1, hi = lane >> 4;
    int n = ntile * 16 + (lane & 15);
    int k = kf * 32 + k_off(v, pair, hi);
    out[idx] = (__bf16)Whr[((long)l * NP + n) * NH + k];
}

__global__ void pack_wfc_k(const float* __restrict__ Wfc, __bf16* __restrict__ out) {
    long idx = (long)blockIdx.x * 256 + threadIdx.x;   // exactly 16,384
    int ev    = (int)(idx & 15);  long t = idx >> 4;
    int lane  = (int)(t & 31);    t >>= 5;
    int kf    = (int)(t & 3);     t >>= 2;
    int ntile = (int)t;           // 0..7
    int v = ev >> 1, pair = ev & 1, hi = lane >> 4;
    int n = ntile * 16 + (lane & 15);
    int k = kf * 32 + k_off(v, pair, hi);
    out[idx] = (__bf16)Wfc[(long)n * NP + k];
}

__global__ void bias_k(const float* __restrict__ bih, const float* __restrict__ bhh,
                       float* __restrict__ out) {
    int idx = blockIdx.x * 256 + threadIdx.x;          // exactly 12,288
    out[idx] = bih[idx] + bhh[idx];
}

__global__ void zero_u4_k(uint4* __restrict__ p, long n) {
    long idx = (long)blockIdx.x * 256 + threadIdx.x;
    if (idx < n) p[idx] = make_uint4(0u, 0u, 0u, 0u);
}

// ---------------------------------------------------------------------------
// Fused LSTMP cell: gates GEMM (bf16 WMMA) + gate math + projection GEMM.
// Grid: 32 blocks (one 16-row batch tile), 256 threads = 8 waves.
// Wave w owns gate columns [w*64, w*64+64) of each of i/f/g/o.
// ---------------------------------------------------------------------------
__launch_bounds__(256, 1)
__global__ void lstm_cell_k(const float*  __restrict__ x_f32,   // null or base+t*P, row stride T*P
                            const __bf16* __restrict__ x_bf16,  // row stride 128
                            const __bf16* __restrict__ h_in,    // [B][128]
                            __bf16*       __restrict__ h_out,   // [B][128]
                            __bf16*       __restrict__ seq_out, // nullable, row stride 128
                            float*        __restrict__ c_st,    // packed [32][32][32][8] f32
                            const __bf16* __restrict__ Wg,      // packed, this layer
                            const __bf16* __restrict__ Whr,     // packed, this layer
                            const float*  __restrict__ bias) {  // [2048]
    __shared__ __align__(16) __bf16 shX[16 * NP];
    __shared__ __align__(16) __bf16 shH[16 * NP];
    __shared__ __align__(16) __bf16 shHF[16 * NH];

    const int tid  = threadIdx.x;
    const int lane = tid & 31;
    const int wv   = tid >> 5;
    const int m0   = blockIdx.x * 16;
    const int hi8  = (lane >> 4) * 8;
    const int lnLo = lane & 15;

    // ---- phase 1: stage X and H tiles (bf16) into LDS ----
    {
        int row = tid >> 4;
        int col = (tid & 15) * 8;
        if (x_f32) {
            const float* src = x_f32 + (long)(m0 + row) * (NT * NP) + col;
            float4 a = *(const float4*)(src);
            float4 b = *(const float4*)(src + 4);
            __bf16* d = &shX[row * NP + col];
            d[0] = (__bf16)a.x; d[1] = (__bf16)a.y; d[2] = (__bf16)a.z; d[3] = (__bf16)a.w;
            d[4] = (__bf16)b.x; d[5] = (__bf16)b.y; d[6] = (__bf16)b.z; d[7] = (__bf16)b.w;
        } else {
            const __bf16* src = x_bf16 + (long)(m0 + row) * NP + col;
            *(uint4*)&shX[row * NP + col] = *(const uint4*)src;
        }
        const __bf16* hs = h_in + (long)(m0 + row) * NP + col;
        *(uint4*)&shH[row * NP + col] = *(const uint4*)hs;
    }
    __syncthreads();

    // ---- phase 2: build A fragments (shared by all gate tiles of this wave) ----
    Frag ax[4], ah[4];
    {
        int rowb = lnLo * NP;
#pragma unroll
        for (int kf = 0; kf < 4; ++kf) {
            const uint4* p = (const uint4*)&shX[rowb + kf * 32 + hi8];
            ax[kf].q[0] = p[0];
            ax[kf].q[1] = p[2];     // +16 bf16 elements = K 16..23
            const uint4* q = (const uint4*)&shH[rowb + kf * 32 + hi8];
            ah[kf].q[0] = q[0];
            ah[kf].q[1] = q[2];
        }
    }

    // ---- phase 3+4: per column sub-tile j, compute i/f/g/o then gate math ----
    const int btile = blockIdx.x;
#pragma unroll
    for (int j = 0; j < 4; ++j) {
        v8f ga[4];
#pragma unroll
        for (int g = 0; g < 4; ++g) {
            int ntile = g * 32 + wv * 4 + j;                 // 0..127
            const __bf16* wb = Wg + (long)ntile * 8 * 512 + lane * 16;
            v8f a = {};
#pragma unroll
            for (int kf = 0; kf < 4; ++kf) {                 // X @ Wih^T
                Frag bf;
                const uint4* p = (const uint4*)(wb + kf * 512);
                bf.q[0] = p[0]; bf.q[1] = p[1];
                a = __builtin_amdgcn_wmma_f32_16x16x32_bf16(
                        false, ax[kf].v, false, bf.v, (short)0, a, false, false);
            }
#pragma unroll
            for (int kf = 0; kf < 4; ++kf) {                 // H @ Whh^T
                Frag bf;
                const uint4* p = (const uint4*)(wb + (kf + 4) * 512);
                bf.q[0] = p[0]; bf.q[1] = p[1];
                a = __builtin_amdgcn_wmma_f32_16x16x32_bf16(
                        false, ah[kf].v, false, bf.v, (short)0, a, false, false);
            }
            float bv = bias[ntile * 16 + lnLo];
#pragma unroll
            for (int r = 0; r < 8; ++r) a[r] += bv;
            ga[g] = a;
        }

        // gate math: c' = sig(f)*c + sig(i)*tanh(g); h_full = sig(o)*tanh(c')
        int htile = wv * 4 + j;                              // 0..31
        float* cp = c_st + (((long)btile * 32 + htile) * 32 + lane) * 8;
        float4 c0 = *(float4*)cp;
        float4 c1 = *(float4*)(cp + 4);
        float cold[8] = {c0.x, c0.y, c0.z, c0.w, c1.x, c1.y, c1.z, c1.w};
        int colb = htile * 16 + lnLo;
#pragma unroll
        for (int r = 0; r < 8; ++r) {
            float cn = sigf(ga[1][r]) * cold[r] + sigf(ga[0][r]) * tanh_f(ga[2][r]);
            cold[r] = cn;
            float hf = sigf(ga[3][r]) * tanh_f(cn);
            shHF[(hi8 + r) * NH + colb] = (__bf16)hf;
        }
        *(float4*)cp       = make_float4(cold[0], cold[1], cold[2], cold[3]);
        *(float4*)(cp + 4) = make_float4(cold[4], cold[5], cold[6], cold[7]);
    }
    __syncthreads();

    // ---- phase 5: projection h = h_full @ Whr^T (M=16, N=128, K=512) ----
    {
        v8f a = {};
        int rowb = lnLo * NH;
        const __bf16* wp = Whr + (long)wv * 16 * 512 + lane * 16;
#pragma unroll
        for (int kf = 0; kf < 16; ++kf) {
            Frag af;
            const uint4* p = (const uint4*)&shHF[rowb + kf * 32 + hi8];
            af.q[0] = p[0]; af.q[1] = p[2];
            Frag bf;
            const uint4* q = (const uint4*)(wp + kf * 512);
            bf.q[0] = q[0]; bf.q[1] = q[1];
            a = __builtin_amdgcn_wmma_f32_16x16x32_bf16(
                    false, af.v, false, bf.v, (short)0, a, false, false);
        }
        int col = wv * 16 + lnLo;
#pragma unroll
        for (int r = 0; r < 8; ++r) {
            __bf16 hv = (__bf16)a[r];
            h_out[(long)(m0 + hi8 + r) * NP + col] = hv;
            if (seq_out) seq_out[(long)(m0 + hi8 + r) * NP + col] = hv;
        }
    }
}

// ---------------------------------------------------------------------------
// Decode output head: out[:, s, :] = sigmoid(h_top) @ Wfc^T + b_fc  (fp32 out)
// ---------------------------------------------------------------------------
__launch_bounds__(256, 1)
__global__ void fc_out_k(const __bf16* __restrict__ htop,
                         const __bf16* __restrict__ WfcP,
                         const float*  __restrict__ bfc,
                         float*        __restrict__ out,
                         int step) {
    __shared__ __align__(16) __bf16 shS[16 * NP];
    const int tid  = threadIdx.x;
    const int lane = tid & 31;
    const int wv   = tid >> 5;
    const int m0   = blockIdx.x * 16;
    const int hi8  = (lane >> 4) * 8;
    const int lnLo = lane & 15;
    {
        int row = tid >> 4;
        int col = (tid & 15) * 8;
        union { uint4 u; __bf16 b[8]; } t;
        t.u = *(const uint4*)(htop + (long)(m0 + row) * NP + col);
        __bf16* d = &shS[row * NP + col];
#pragma unroll
        for (int e = 0; e < 8; ++e) d[e] = (__bf16)sigf((float)t.b[e]);
    }
    __syncthreads();

    v8f a = {};
    int rowb = lnLo * NP;
    const __bf16* wp = WfcP + (long)wv * 4 * 512 + lane * 16;
#pragma unroll
    for (int kf = 0; kf < 4; ++kf) {
        Frag af;
        const uint4* p = (const uint4*)&shS[rowb + kf * 32 + hi8];
        af.q[0] = p[0]; af.q[1] = p[2];
        Frag bf;
        const uint4* q = (const uint4*)(wp + kf * 512);
        bf.q[0] = q[0]; bf.q[1] = q[1];
        a = __builtin_amdgcn_wmma_f32_16x16x32_bf16(
                false, af.v, false, bf.v, (short)0, a, false, false);
    }
    int col = wv * 16 + lnLo;
    float bv = bfc[col];
#pragma unroll
    for (int r = 0; r < 8; ++r) {
        out[((long)(m0 + hi8 + r) * NSTEP + step) * NP + col] = a[r] + bv;
    }
}

// ---------------------------------------------------------------------------
extern "C" void kernel_launch(void* const* d_in, const int* in_sizes, int n_in,
                              void* d_out, int out_size, void* d_ws, size_t ws_size,
                              hipStream_t stream) {
    (void)in_sizes; (void)n_in; (void)out_size; (void)ws_size;
    const float* x   = (const float*)d_in[0];   // [B,T,P]
    const float* Wih = (const float*)d_in[1];   // [L,4H,P]
    const float* Whh = (const float*)d_in[2];   // [L,4H,P]
    const float* bih = (const float*)d_in[3];   // [L,4H]
    const float* bhh = (const float*)d_in[4];   // [L,4H]
    const float* Whr = (const float*)d_in[5];   // [L,P,H]
    const float* Wfc = (const float*)d_in[6];   // [P,P]
    const float* bfc = (const float*)d_in[7];   // [P]
    float* out = (float*)d_out;

    char* ws = (char*)d_ws;
    __bf16* seq   = (__bf16*)(ws + OFF_SEQ);
    __bf16* wg    = (__bf16*)(ws + OFF_WG);
    __bf16* whrp  = (__bf16*)(ws + OFF_WHR);
    __bf16* wfcp  = (__bf16*)(ws + OFF_WFC);
    float*  biasf = (float*)(ws + OFF_BIAS);
    __bf16* hst   = (__bf16*)(ws + OFF_H);
    float*  cst   = (float*)(ws + OFF_C);
    __bf16* zb    = (__bf16*)(ws + OFF_ZERO);

    // one-time prep (re-run every call for determinism)
    zero_u4_k<<<1760, 256, 0, stream>>>((uint4*)(ws + OFF_H), 450560L);
    pack_gates_k<<<12288, 256, 0, stream>>>(Wih, Whh, wg);
    pack_whr_k<<<1536, 256, 0, stream>>>(Whr, whrp);
    pack_wfc_k<<<64, 256, 0, stream>>>(Wfc, wfcp);
    bias_k<<<48, 256, 0, stream>>>(bih, bhh, biasf);

    // ---- encode: layer-by-layer scan over time; seq buffer updated in place ----
    for (int l = 0; l < NL; ++l) {
        __bf16* hl = hst + (long)l * NB * NP;
        float*  cl = cst + (long)l * NB * NH;
        const __bf16* wgl  = wg   + (long)l * 524288;
        const __bf16* whrl = whrp + (long)l * 65536;
        const float*  bl   = biasf + (long)l * NG;
        for (int t = 0; t < NT; ++t) {
            const float*  xf = (l == 0) ? (x + (long)t * NP) : nullptr;
            const __bf16* xb = (l == 0) ? nullptr : (seq + (long)t * NB * NP);
            lstm_cell_k<<<32, 256, 0, stream>>>(xf, xb, hl, hl,
                                                seq + (long)t * NB * NP,
                                                cl, wgl, whrl, bl);
        }
    }

    // ---- decode: 32 autoregressive steps, zero input to layer 0 ----
    for (int s = 0; s < NSTEP; ++s) {
        for (int l = 0; l < NL; ++l) {
            const __bf16* xb = (l == 0) ? zb : (hst + (long)(l - 1) * NB * NP);
            __bf16* hl = hst + (long)l * NB * NP;
            float*  cl = cst + (long)l * NB * NH;
            lstm_cell_k<<<32, 256, 0, stream>>>(nullptr, xb, hl, hl, nullptr,
                                                cl,
                                                wg   + (long)l * 524288,
                                                whrp + (long)l * 65536,
                                                biasf + (long)l * NG);
        }
        fc_out_k<<<32, 256, 0, stream>>>(hst + (long)5 * NB * NP, wfcp, bfc, out, s);
    }
}
